// HeteroRGCNLayer_25202868093363
// MI455X (gfx1250) — matrix operand order repaired and checked by reading, not compiled, not run on hardware
//
#include <hip/hip_runtime.h>
#include <hip/hip_bf16.h>

#define N_NODES 100000
#define DIM     128
#define NEDGE   800000
#define NETYPE  2
#define NHEAD   4
#define HDIM    32
#define EPSLN   1e-5f

typedef __attribute__((ext_vector_type(16))) __bf16 v16bf;
typedef __attribute__((ext_vector_type(8)))  float  v8f;

union FragU { uint4 u[2]; v16bf v; };

__device__ __forceinline__ unsigned short f2bf(float f) {
  unsigned u = __float_as_uint(f);
  unsigned r = u + 0x7FFFu + ((u >> 16) & 1u);   // round-to-nearest-even
  return (unsigned short)(r >> 16);
}
__device__ __forceinline__ float bf2f(unsigned short h) {
  return __uint_as_float(((unsigned)h) << 16);
}

// ---------------------------------------------------------------------------
// Pre-swizzle all 8 weight matrices [128(K) x 128(N)] f32 -> bf16 WMMA B-frag
// layout: frag[(e*4+w)*16384 + (n_tile*4+k_step)*512 + lane*16 + j]
//   K = k_step*32 + ((lane>=16)?16:0) + j ,  N = n_tile*16 + (lane&15)
// ---------------------------------------------------------------------------
__global__ __launch_bounds__(256) void wfrag_kernel(
    const float* __restrict__ Wt, const float* __restrict__ Kw,
    const float* __restrict__ Qw, const float* __restrict__ Vw,
    unsigned short* __restrict__ frag) {
  int t = blockIdx.x * blockDim.x + threadIdx.x;   // 0 .. 8*16384-1
  int wi = t >> 14;            // e*4 + w
  int e  = wi >> 2, w = wi & 3;
  int r  = t & 16383;
  int fr = r >> 9;             // n_tile*4 + k_step
  int lane = (r >> 4) & 31;
  int j  = r & 15;
  int n_tile = fr >> 2, k_step = fr & 3;
  int K  = k_step * 32 + ((lane >> 4) << 4) + j;
  int Nc = n_tile * 16 + (lane & 15);
  const float* W = (w == 0) ? Wt : (w == 1) ? Kw : (w == 2) ? Qw : Vw;
  frag[t] = f2bf(W[(size_t)e * DIM * DIM + (size_t)K * DIM + Nc]);
}

// ---------------------------------------------------------------------------
// Fused: Wh0 = feat@Wt+bt ; k,q,v = Wh0@{Kw,Qw,Vw}+b ; 4x4 cross-head attn ;
// writes Wh (bf16) rows for one tile of 16 nodes.  256 thr = 8 waves,
// wave n owns output columns [16n,16n+16).
// ---------------------------------------------------------------------------
__global__ __launch_bounds__(256) void gemm_attn_kernel(
    const float* __restrict__ feat,
    const unsigned short* __restrict__ wfrag,   // this etype: 4*16384 bf16 frags
    const float* __restrict__ bt, const float* __restrict__ Kb,
    const float* __restrict__ Qb, const float* __restrict__ Vb,
    unsigned short* __restrict__ Wh) {
  __shared__ __align__(16) unsigned short sFeat[16 * DIM];
  __shared__ __align__(16) unsigned short sWh[16 * DIM];
  __shared__ float sK[16 * DIM];
  __shared__ float sQ[16 * DIM];
  __shared__ float sV[16 * DIM];
  __shared__ float sS[16 * 16];
  __shared__ float sP[16 * 16];

  const int t    = threadIdx.x;
  const int wave = t >> 5;          // n_tile (0..7)
  const int lane = t & 31;
  const int m0   = blockIdx.x * 16;

  // load 16x128 f32 feat tile -> bf16 LDS (each thread converts 8 elements)
  {
    int row = t >> 4, cb = (t & 15) * 8;
    const float4* gp = (const float4*)(feat + (size_t)(m0 + row) * DIM + cb);
    float4 a = gp[0], b = gp[1];
    unsigned short h[8] = { f2bf(a.x), f2bf(a.y), f2bf(a.z), f2bf(a.w),
                            f2bf(b.x), f2bf(b.y), f2bf(b.z), f2bf(b.w) };
    *(uint4*)(&sFeat[row * DIM + cb]) = *(uint4*)h;
  }
  __syncthreads();

  const int am     = lane & 15;          // A: row this lane supplies
  const int akb    = (lane >> 4) * 16;   // A: K sub-offset
  const int cn     = wave * 16 + (lane & 15); // C/D: column
  const int cmBase = (lane >> 4) * 8;    // C/D: row base

  // ---- GEMM1: Wh0 = feat @ Wt + bt -> sWh (bf16) ----
  {
    float bv = bt[cn];
    v8f acc = { bv, bv, bv, bv, bv, bv, bv, bv };
    const unsigned short* Bp = wfrag + (size_t)(wave * 4) * 512 + lane * 16;
#pragma unroll
    for (int ks = 0; ks < 4; ++ks) {
      FragU A, B;
      const unsigned short* ap = &sFeat[am * DIM + ks * 32 + akb];
      A.u[0] = *(const uint4*)ap;
      A.u[1] = *(const uint4*)(ap + 8);
      const unsigned short* bp = Bp + ks * 512;
      B.u[0] = *(const uint4*)bp;
      B.u[1] = *(const uint4*)(bp + 8);
      acc = __builtin_amdgcn_wmma_f32_16x16x32_bf16(false, A.v, false, B.v,
                                                    (short)0, acc, false, false);
    }
#pragma unroll
    for (int v = 0; v < 8; ++v) sWh[(cmBase + v) * DIM + cn] = f2bf(acc[v]);
  }
  __syncthreads();

  // ---- GEMM2-4: k,q,v = Wh0 @ {Kw,Qw,Vw} + bias (A-frags shared) ----
  {
    float kv = Kb[cn], qv = Qb[cn], vv = Vb[cn];
    v8f accK = { kv, kv, kv, kv, kv, kv, kv, kv };
    v8f accQ = { qv, qv, qv, qv, qv, qv, qv, qv };
    v8f accV = { vv, vv, vv, vv, vv, vv, vv, vv };
    const unsigned short* BpK = wfrag + 1 * 16384 + (size_t)(wave * 4) * 512 + lane * 16;
    const unsigned short* BpQ = wfrag + 2 * 16384 + (size_t)(wave * 4) * 512 + lane * 16;
    const unsigned short* BpV = wfrag + 3 * 16384 + (size_t)(wave * 4) * 512 + lane * 16;
#pragma unroll
    for (int ks = 0; ks < 4; ++ks) {
      FragU A, B;
      const unsigned short* ap = &sWh[am * DIM + ks * 32 + akb];
      A.u[0] = *(const uint4*)ap;
      A.u[1] = *(const uint4*)(ap + 8);
      const unsigned short* bp = BpK + ks * 512;
      B.u[0] = *(const uint4*)bp; B.u[1] = *(const uint4*)(bp + 8);
      accK = __builtin_amdgcn_wmma_f32_16x16x32_bf16(false, A.v, false, B.v,
                                                     (short)0, accK, false, false);
      bp = BpQ + ks * 512;
      B.u[0] = *(const uint4*)bp; B.u[1] = *(const uint4*)(bp + 8);
      accQ = __builtin_amdgcn_wmma_f32_16x16x32_bf16(false, A.v, false, B.v,
                                                     (short)0, accQ, false, false);
      bp = BpV + ks * 512;
      B.u[0] = *(const uint4*)bp; B.u[1] = *(const uint4*)(bp + 8);
      accV = __builtin_amdgcn_wmma_f32_16x16x32_bf16(false, A.v, false, B.v,
                                                     (short)0, accV, false, false);
    }
#pragma unroll
    for (int v = 0; v < 8; ++v) {
      int m = cmBase + v;
      sK[m * DIM + cn] = accK[v];
      sQ[m * DIM + cn] = accQ[v];
      sV[m * DIM + cn] = accV[v];
    }
  }
  __syncthreads();

  // ---- attention scores: one (node,h,g) per thread ----
  {
    int node = t >> 4;
    int h = (t >> 2) & 3;
    int g = t & 3;
    const float* q = &sQ[node * DIM + h * HDIM];
    const float* k = &sK[node * DIM + g * HDIM];
    float s = 0.f;
#pragma unroll
    for (int d = 0; d < HDIM; ++d) s += q[d] * k[d];
    sS[node * 16 + h * 4 + g] = s * 0.17677669529663687f; // 1/sqrt(32)
  }
  __syncthreads();
  if (t < 64) { // softmax over g for (node,h)
    int node = t >> 2, h = t & 3;
    float* s = &sS[node * 16 + h * 4];
    float mx = fmaxf(fmaxf(s[0], s[1]), fmaxf(s[2], s[3]));
    float e0 = __expf(s[0] - mx), e1 = __expf(s[1] - mx);
    float e2 = __expf(s[2] - mx), e3 = __expf(s[3] - mx);
    float inv = 1.f / (e0 + e1 + e2 + e3);
    float* p = &sP[node * 16 + h * 4];
    p[0] = e0 * inv; p[1] = e1 * inv; p[2] = e2 * inv; p[3] = e3 * inv;
  }
  __syncthreads();

  // ---- Wh = a @ v, store bf16 rows (8 elems/thread) ----
  {
    int node = t >> 4, part = t & 15;
    int d0 = part * 8;
    unsigned short out[8];
#pragma unroll
    for (int i = 0; i < 8; ++i) {
      int d = d0 + i;
      int h = d >> 5, off = d & 31;
      const float* p = &sP[node * 16 + h * 4];
      const float* v = &sV[node * DIM + off];
      out[i] = f2bf(p[0] * v[0 * HDIM] + p[1] * v[1 * HDIM] +
                    p[2] * v[2 * HDIM] + p[3] * v[3 * HDIM]);
    }
    *(uint4*)(Wh + (size_t)(m0 + node) * DIM + d0) = *(uint4*)out;
  }
}

// ---------------------------------------------------------------------------
// One wave per edge (grid-stride): gather bf16 row Wh[src], f32 atomicAdd
// into acc[dst]; lane 0 counts degree.
// ---------------------------------------------------------------------------
__global__ __launch_bounds__(256) void scatter_kernel(
    const unsigned short* __restrict__ Wh,
    const int* __restrict__ src, const int* __restrict__ dst,
    float* __restrict__ acc, float* __restrict__ deg) {
  int lane = threadIdx.x & 31;
  int wid  = (blockIdx.x * blockDim.x + threadIdx.x) >> 5;
  int nw   = (gridDim.x * blockDim.x) >> 5;
  for (int e = wid; e < NEDGE; e += nw) {
    int s = src[e], d = dst[e];
    uint2 raw = *(const uint2*)(Wh + (size_t)s * DIM + lane * 4);
    float f0 = bf2f((unsigned short)(raw.x & 0xffff));
    float f1 = bf2f((unsigned short)(raw.x >> 16));
    float f2 = bf2f((unsigned short)(raw.y & 0xffff));
    float f3 = bf2f((unsigned short)(raw.y >> 16));
    float* ap = acc + (size_t)d * DIM + lane * 4;
    atomicAdd(ap + 0, f0);
    atomicAdd(ap + 1, f1);
    atomicAdd(ap + 2, f2);
    atomicAdd(ap + 3, f3);
    if (lane == 0) atomicAdd(deg + d, 1.0f);
  }
}

// ---------------------------------------------------------------------------
// One wave per node: h = feat + sum_e acc_e/max(deg_e,1); LayerNorm -> out.
// ---------------------------------------------------------------------------
__global__ __launch_bounds__(256) void finalize_kernel(
    const float* __restrict__ feat,
    const float* __restrict__ acc,   // [2][N][128]
    const float* __restrict__ deg,   // [2][N]
    const float* __restrict__ ln_g, const float* __restrict__ ln_b,
    float* __restrict__ out) {
  int lane = threadIdx.x & 31;
  int row  = (blockIdx.x * blockDim.x + threadIdx.x) >> 5;
  if (row >= N_NODES) return;
  size_t base = (size_t)row * DIM + lane * 4;
  float4 f  = *(const float4*)(feat + base);
  float4 a0 = *(const float4*)(acc + base);
  float4 a1 = *(const float4*)(acc + (size_t)N_NODES * DIM + base);
  float id0 = 1.f / fmaxf(deg[row], 1.f);
  float id1 = 1.f / fmaxf(deg[N_NODES + row], 1.f);
  float h[4];
  h[0] = f.x + a0.x * id0 + a1.x * id1;
  h[1] = f.y + a0.y * id0 + a1.y * id1;
  h[2] = f.z + a0.z * id0 + a1.z * id1;
  h[3] = f.w + a0.w * id0 + a1.w * id1;
  float ps = h[0] + h[1] + h[2] + h[3];
#pragma unroll
  for (int off = 16; off > 0; off >>= 1) ps += __shfl_xor(ps, off, 32);
  float mu = ps * (1.0f / DIM);
  float pv = 0.f;
#pragma unroll
  for (int i = 0; i < 4; ++i) { float dd = h[i] - mu; pv += dd * dd; }
#pragma unroll
  for (int off = 16; off > 0; off >>= 1) pv += __shfl_xor(pv, off, 32);
  float rstd = rsqrtf(pv * (1.0f / DIM) + EPSLN);
  float4 g = *(const float4*)(ln_g + lane * 4);
  float4 b = *(const float4*)(ln_b + lane * 4);
  float4 o;
  o.x = (h[0] - mu) * rstd * g.x + b.x;
  o.y = (h[1] - mu) * rstd * g.y + b.y;
  o.z = (h[2] - mu) * rstd * g.z + b.z;
  o.w = (h[3] - mu) * rstd * g.w + b.w;
  *(float4*)(out + base) = o;
}

extern "C" void kernel_launch(void* const* d_in, const int* in_sizes, int n_in,
                              void* d_out, int out_size, void* d_ws, size_t ws_size,
                              hipStream_t stream) {
  (void)in_sizes; (void)n_in; (void)out_size; (void)ws_size;
  const float* feat = (const float*)d_in[0];
  const int*   src  = (const int*)d_in[1];
  const int*   dst  = (const int*)d_in[2];
  const float* Wt   = (const float*)d_in[3];
  const float* bt   = (const float*)d_in[4];
  const float* Kw   = (const float*)d_in[5];
  const float* Kb   = (const float*)d_in[6];
  const float* Qw   = (const float*)d_in[7];
  const float* Qb   = (const float*)d_in[8];
  const float* Vw   = (const float*)d_in[9];
  const float* Vb   = (const float*)d_in[10];
  const float* ln_g = (const float*)d_in[11];
  const float* ln_b = (const float*)d_in[12];
  float* out = (float*)d_out;

  // workspace layout
  char* ws = (char*)d_ws;
  unsigned short* wfrag = (unsigned short*)ws;                       // 8*16384 bf16
  unsigned short* Wh    = (unsigned short*)(ws + 262144);            // N*128 bf16
  float* acc = (float*)(ws + 262144 + (size_t)N_NODES * DIM * 2);    // [2][N][128] f32
  float* deg = (float*)((char*)acc + (size_t)2 * N_NODES * DIM * 4); // [2][N] f32

  // zero acc + deg (contiguous)
  hipMemsetAsync(acc, 0, (size_t)2 * N_NODES * DIM * 4 + (size_t)2 * N_NODES * 4,
                 stream);

  // pre-swizzle weights into WMMA B-frag layout
  wfrag_kernel<<<512, 256, 0, stream>>>(Wt, Kw, Qw, Vw, wfrag);

  for (int e = 0; e < NETYPE; ++e) {
    gemm_attn_kernel<<<N_NODES / 16, 256, 0, stream>>>(
        feat, wfrag + (size_t)e * 4 * 16384,
        bt + e * DIM, Kb + e * DIM, Qb + e * DIM, Vb + e * DIM, Wh);
    scatter_kernel<<<2048, 256, 0, stream>>>(
        Wh, src + (size_t)e * NEDGE, dst + (size_t)e * NEDGE,
        acc + (size_t)e * N_NODES * DIM, deg + (size_t)e * N_NODES);
  }

  finalize_kernel<<<(N_NODES * 32 + 255) / 256, 256, 0, stream>>>(
      feat, acc, deg, ln_g, ln_b, out);
}
